// ImplicitH2Projection_10634339025029
// MI455X (gfx1250) — matrix operand order repaired and checked by reading, not compiled, not run on hardware
//
#include <hip/hip_runtime.h>

typedef __bf16 bf16_t;
typedef __attribute__((ext_vector_type(16))) __bf16 v16bf;
typedef __attribute__((ext_vector_type(8)))  __bf16 v8bf;
typedef __attribute__((ext_vector_type(8)))  float   v8f;

static __device__ __forceinline__ v16bf cat16(v8bf lo, v8bf hi) {
  return __builtin_shufflevector(lo, hi, 0,1,2,3,4,5,6,7,8,9,10,11,12,13,14,15);
}

// ---------------------------------------------------------------------------
// bf16 WMMA GEMM:  D = alpha * A @ B + beta * Csrc
//   A   : M x K row-major bf16
//   Bt  : N x K row-major bf16  (B transposed -> contiguous K per column)
//   Csrc: M x N row-major f32 (may be null when beta == 0)
//   D   : M x N row-major, f32 or bf16 (template)
// Block tile 128x128, K-slab 32, 256 threads = 8 waves, each wave 64x32.
// Double-buffered LDS with register prefetch: next slab's global loads are
// issued before the 8 WMMAs so the loadcnt wait lands after the math.
// M, N multiples of 128; K multiple of 32.
// ---------------------------------------------------------------------------
template <bool OUT_BF16>
__global__ __launch_bounds__(256)
void gemm_bf16_wmma(const bf16_t* __restrict__ A,
                    const bf16_t* __restrict__ Bt,
                    const float*  __restrict__ Csrc,
                    void*         __restrict__ D,
                    int M, int N, int K,
                    float alpha, float beta)
{
  constexpr int BM = 128, BN = 128, BK = 32;
  constexpr int LDSS = 40;                  // padded row stride (bf16) = 80B
  __shared__ bf16_t Asm[2][BM * LDSS];
  __shared__ bf16_t Bsm[2][BN * LDSS];

  const int tid  = threadIdx.x;
  const int wave = tid >> 5;
  const int lane = tid & 31;
  const int l15  = lane & 15;
  const int lh   = lane >> 4;               // half-wave select
  const int mWave = (wave >> 2) * 64;       // 2 wave rows
  const int nWave = (wave & 3) * 32;        // 4 wave cols

  const int blockM = blockIdx.y * BM;
  const int blockN = blockIdx.x * BN;

  // Fixed per-thread staging coordinates: rows r and r+64, 8 bf16 at col c.
  const int r = tid >> 2;                   // 0..63
  const int c = (tid & 3) * 8;              // 0,8,16,24
  const bf16_t* gA0 = A  + (size_t)(blockM + r)      * K + c;
  const bf16_t* gA1 = A  + (size_t)(blockM + r + 64) * K + c;
  const bf16_t* gB0 = Bt + (size_t)(blockN + r)      * K + c;
  const bf16_t* gB1 = Bt + (size_t)(blockN + r + 64) * K + c;
  const int sOff0 = r * LDSS + c;
  const int sOff1 = (r + 64) * LDSS + c;

  v8f acc[4][2];
#pragma unroll
  for (int i = 0; i < 4; ++i)
#pragma unroll
    for (int j = 0; j < 2; ++j)
      acc[i][j] = (v8f){0.f,0.f,0.f,0.f,0.f,0.f,0.f,0.f};

  // Prologue: stage slab 0 into buffer 0.
  {
    v8bf a0 = *(const v8bf*)gA0;
    v8bf a1 = *(const v8bf*)gA1;
    v8bf b0 = *(const v8bf*)gB0;
    v8bf b1 = *(const v8bf*)gB1;
    *(v8bf*)(&Asm[0][sOff0]) = a0;
    *(v8bf*)(&Asm[0][sOff1]) = a1;
    *(v8bf*)(&Bsm[0][sOff0]) = b0;
    *(v8bf*)(&Bsm[0][sOff1]) = b1;
  }
  __syncthreads();

  int buf = 0;
  for (int k0 = 0; k0 < K; k0 += BK) {
    const bool hasNext = (k0 + BK) < K;
    v8bf nA0, nA1, nB0, nB1;
    if (hasNext) {
      // Issue next slab's global loads now; waited on only at the stores.
      nA0 = *(const v8bf*)(gA0 + k0 + BK);
      nA1 = *(const v8bf*)(gA1 + k0 + BK);
      nB0 = *(const v8bf*)(gB0 + k0 + BK);
      nB1 = *(const v8bf*)(gB1 + k0 + BK);
      if (k0 + 2 * BK < K) {               // warm WGP$ for slab after next
        __builtin_prefetch(gA0 + k0 + 2 * BK, 0, 3);
        __builtin_prefetch(gB0 + k0 + 2 * BK, 0, 3);
      }
    }

    // B fragments: lane -> col n = base + l15, K = 8*lh+(0..7) and +16..+23.
    v16bf bfrag[2];
#pragma unroll
    for (int j = 0; j < 2; ++j) {
      const bf16_t* p = &Bsm[buf][(nWave + j * 16 + l15) * LDSS + lh * 8];
      bfrag[j] = cat16(*(const v8bf*)p, *(const v8bf*)(p + 16));
    }
    // A fragments: lane -> row m = base + l15, K = 8*lh+(0..7) and +16..+23.
#pragma unroll
    for (int i = 0; i < 4; ++i) {
      const bf16_t* p = &Asm[buf][(mWave + i * 16 + l15) * LDSS + lh * 8];
      v16bf afrag = cat16(*(const v8bf*)p, *(const v8bf*)(p + 16));
#pragma unroll
      for (int j = 0; j < 2; ++j) {
        acc[i][j] = __builtin_amdgcn_wmma_f32_16x16x32_bf16(
            false, afrag, false, bfrag[j], (short)0, acc[i][j], false, false);
      }
    }

    if (hasNext) {
      const int nb = buf ^ 1;
      *(v8bf*)(&Asm[nb][sOff0]) = nA0;
      *(v8bf*)(&Asm[nb][sOff1]) = nA1;
      *(v8bf*)(&Bsm[nb][sOff0]) = nB0;
      *(v8bf*)(&Bsm[nb][sOff1]) = nB1;
      __syncthreads();
      buf = nb;
    }
  }

  // Epilogue. C/D layout: lane -> N = l15, M = 8*lh + v (v = VGPR index).
#pragma unroll
  for (int i = 0; i < 4; ++i) {
#pragma unroll
    for (int j = 0; j < 2; ++j) {
      const int mBase = blockM + mWave + i * 16 + lh * 8;
      const int n     = blockN + nWave + j * 16 + l15;
#pragma unroll
      for (int v = 0; v < 8; ++v) {
        const size_t idx = (size_t)(mBase + v) * N + n;
        float val = alpha * acc[i][j][v];
        if (beta != 0.f) val += beta * Csrc[idx];
        if (OUT_BF16) ((bf16_t*)D)[idx] = (bf16_t)val;
        else          ((float*) D)[idx] = val;
      }
    }
  }
}

// Tiled transpose + f32->bf16 convert: dst[n][m] = (bf16)src[m][n].
// src: M x N f32; dst: N x M bf16. Grid (N/32, M/32), block (32,8).
__global__ void transpose_cvt(const float* __restrict__ src,
                              bf16_t* __restrict__ dst, int M, int N)
{
  __shared__ float tile[32][33];
  const int bx = blockIdx.x * 32;  // N
  const int by = blockIdx.y * 32;  // M
  const int tx = threadIdx.x, ty = threadIdx.y;
#pragma unroll
  for (int i = 0; i < 32; i += 8)
    tile[ty + i][tx] = src[(size_t)(by + ty + i) * N + bx + tx];
  __syncthreads();
#pragma unroll
  for (int i = 0; i < 32; i += 8)
    dst[(size_t)(bx + ty + i) * M + by + tx] = (bf16_t)tile[tx][ty + i];
}

// u[j] = sum_i C[i][j]   (column sums)
__global__ void colsum_kernel(const float* __restrict__ Cmat,
                              float* __restrict__ u, int p, int n)
{
  const int j = blockIdx.x * blockDim.x + threadIdx.x;
  if (j >= n) return;
  float s = 0.f;
  for (int i = 0; i < p; ++i) s += Cmat[(size_t)i * n + j];
  u[j] = s;
}

__global__ void copy_vec(const float* __restrict__ src,
                         float* __restrict__ dst, int n)
{
  const int i = blockIdx.x * blockDim.x + threadIdx.x;
  if (i < n) dst[i] = src[i];
}

// vout[i] += sum_{j in chunk} A[j][i] * vin[j]  (i.e. A^T @ vin, split over j)
__global__ void matvecT_acc(const float* __restrict__ A,
                            const float* __restrict__ vin,
                            float* __restrict__ vout, int n)
{
  const int i  = blockIdx.x * 256 + threadIdx.x;
  const int j0 = blockIdx.y * 128;
  float s = 0.f;
#pragma unroll 4
  for (int j = j0; j < j0 + 128; ++j) s += A[(size_t)j * n + i] * vin[j];
  atomicAdd(&vout[i], s);
}

// w[i] = sum_j C[i][j] * v[j]; one block per output row.
__global__ void rowdot_kernel(const float* __restrict__ Cmat,
                              const float* __restrict__ v,
                              float* __restrict__ w, int n)
{
  __shared__ float red[256];
  const int i = blockIdx.x;
  float s = 0.f;
  for (int j = threadIdx.x; j < n; j += 256)
    s += Cmat[(size_t)i * n + j] * v[j];
  red[threadIdx.x] = s;
  __syncthreads();
  for (int o = 128; o > 0; o >>= 1) {
    if (threadIdx.x < o) red[threadIdx.x] += red[threadIdx.x + o];
    __syncthreads();
  }
  if (threadIdx.x == 0) w[i] = red[0];
}

// D_hat[i][j] = 0.5*(1 - w[i])/p + 0.5*D_F[i][j]
__global__ void dhat_kernel(const float* __restrict__ w,
                            const float* __restrict__ DF,
                            float* __restrict__ out, int p, int m)
{
  const int idx = blockIdx.x * blockDim.x + threadIdx.x;
  if (idx >= p * m) return;
  const int i = idx / m;
  out[idx] = 0.5f * (1.f - w[i]) / (float)p + 0.5f * DF[idx];
}

extern "C" void kernel_launch(void* const* d_in, const int* in_sizes, int n_in,
                              void* d_out, int out_size, void* d_ws, size_t ws_size,
                              hipStream_t stream)
{
  (void)in_sizes; (void)n_in; (void)out_size; (void)ws_size;
  constexpr int n = 2048, nF = 2048, p = 512, m = 512;
  constexpr int N_SYLV = 49;   // X_1 = RHS precomputed, +49 -> X_50
  constexpr int N_NEUM = 64;   // Neumann iterations for (I - A^T)^{-1}

  const float* Cm  = (const float*)d_in[0];
  const float* A   = (const float*)d_in[1];
  const float* A_F = (const float*)d_in[2];
  const float* B_F = (const float*)d_in[3];
  const float* C_F = (const float*)d_in[4];
  const float* D_F = (const float*)d_in[5];
  float* B_hat = (float*)d_out;                         // n  x m
  float* D_hat = (float*)d_out + (size_t)n * m;         // p  x m

  // Workspace carve-up (256B aligned).
  char* wsp = (char*)d_ws;
  auto carve = [&](size_t bytes) {
    char* r = wsp; wsp += (bytes + 255) & ~(size_t)255; return r;
  };
  float*  RHS = (float*) carve((size_t)nF * n * 4);   // C_F^T C          (f32)
  float*  X   = (float*) carve((size_t)nF * n * 4);   // iterate          (f32)
  bf16_t* Tbf = (bf16_t*)carve((size_t)nF * n * 2);   // A_F^T X          (bf16)
  bf16_t* Xt  = (bf16_t*)carve((size_t)n * nF * 2);   // X^T              (bf16)
  bf16_t* AFt = (bf16_t*)carve((size_t)nF * nF * 2);  // A_F^T            (bf16)
  bf16_t* At  = (bf16_t*)carve((size_t)n * n * 2);    // A^T              (bf16)
  bf16_t* CFt = (bf16_t*)carve((size_t)nF * p * 2);   // C_F^T            (bf16)
  bf16_t* Ct  = (bf16_t*)carve((size_t)n * p * 2);    // C^T              (bf16)
  bf16_t* BFt = (bf16_t*)carve((size_t)m * nF * 2);   // B_F^T            (bf16)
  float*  u   = (float*) carve(n * 4);
  float*  v0  = (float*) carve(n * 4);
  float*  v1  = (float*) carve(n * 4);
  float*  wv  = (float*) carve(p * 4);

  const dim3 tb(32, 8);
  // One-time transposed bf16 operand copies.
  transpose_cvt<<<dim3(nF / 32, nF / 32), tb, 0, stream>>>(A_F, AFt, nF, nF);
  transpose_cvt<<<dim3(n  / 32, n  / 32), tb, 0, stream>>>(A,   At,  n,  n);
  transpose_cvt<<<dim3(nF / 32, p  / 32), tb, 0, stream>>>(C_F, CFt, p,  nF);
  transpose_cvt<<<dim3(n  / 32, p  / 32), tb, 0, stream>>>(Cm,  Ct,  p,  n);
  transpose_cvt<<<dim3(m  / 32, nF / 32), tb, 0, stream>>>(B_F, BFt, nF, m);

  // RHS = C_F^T @ C : M=nF, N=n, K=p
  gemm_bf16_wmma<false><<<dim3(n / 128, nF / 128), 256, 0, stream>>>(
      CFt, Ct, nullptr, RHS, nF, n, p, 1.f, 0.f);

  // X_1 = RHS; keep transposed bf16 copy for the next GEMM's B operand.
  transpose_cvt<<<dim3(n / 32, nF / 32), tb, 0, stream>>>(RHS, Xt, nF, n);

  for (int it = 0; it < N_SYLV; ++it) {
    // T = A_F^T @ X : M=nF, N=n, K=nF  (bf16 output -> next A operand)
    gemm_bf16_wmma<true><<<dim3(n / 128, nF / 128), 256, 0, stream>>>(
        AFt, Xt, nullptr, Tbf, nF, n, nF, 1.f, 0.f);
    // X = T @ A + RHS : M=nF, N=n, K=n
    gemm_bf16_wmma<false><<<dim3(n / 128, nF / 128), 256, 0, stream>>>(
        Tbf, At, RHS, X, nF, n, n, 1.f, 1.f);
    // Xt = (X)^T as bf16
    transpose_cvt<<<dim3(n / 32, nF / 32), tb, 0, stream>>>(X, Xt, nF, n);
  }

  // B_hat = -(Q12^T @ B_F) : M=n, N=m, K=nF ; A-op = Xt = Q12^T
  gemm_bf16_wmma<false><<<dim3(m / 128, n / 128), 256, 0, stream>>>(
      Xt, BFt, nullptr, B_hat, n, m, nF, -1.f, 0.f);

  // ---- D_hat via row sums: rowsum_i = 1 - (C (I-A^T)^{-1} C^T 1)_i ----
  colsum_kernel<<<n / 256, 256, 0, stream>>>(Cm, u, p, n);
  copy_vec<<<n / 256, 256, 0, stream>>>(u, v0, n);
  float* vcur = v0; float* vnext = v1;
  for (int it = 0; it < N_NEUM; ++it) {
    copy_vec<<<n / 256, 256, 0, stream>>>(u, vnext, n);          // vnext = u
    matvecT_acc<<<dim3(n / 256, n / 128), 256, 0, stream>>>(A, vcur, vnext, n);
    float* t = vcur; vcur = vnext; vnext = t;
  }
  rowdot_kernel<<<p, 256, 0, stream>>>(Cm, vcur, wv, n);
  dhat_kernel<<<(p * m) / 256, 256, 0, stream>>>(wv, D_F, D_hat, p, m);
}
